// BitResidualBlock_35605278884131
// MI455X (gfx1250) — matrix operand order repaired and checked by reading, not compiled, not run on hardware
//
#include <hip/hip_runtime.h>

// MI455X / gfx1250, wave32. Conv-as-GEMM on v_wmma_f32_16x16x32_f16.
// All activation staging is CDNA5 async global->LDS b128 DMA (ASYNCcnt),
// double-buffered so the copy for cin-chunk k+1 overlaps chunk k's WMMAs.
//
// Roofline: 6 convs x 51.5 GFLOP = 309 GFLOP, all tensors L2-resident
// (<=64MB each vs 192MB L2; HBM ~128MB => ~5.5us @ 23.3TB/s, negligible).
// Second-order cost is weight re-reads from L2: with a 128x128 block tile
// each conv re-reads ~0.8GB of packed weights (vs 1.6GB at N=64), so we use
// block = 128 Cout x 128 T, wave = 32 x 64 (acc[2][4]), 24 WMMA / chunk.
//
// Precision: ternary weights {-1,0,1} are exact in f16 (per-tensor absmean
// scale folded into the fp32 epilogue); activations f16; accumulate f32.
//
// Layouts: d_in x / final d_out are [B][C][T] f32 (reference). Internals:
//   xf16 [B][T][C] f16 : conv1 input (x transposed once; then conv2 epilogue
//                        emits the next stage's f16 copy directly)
//   h    [B][T][C] f16 : conv1 -> conv2
//   x1/x2 [B][T][C] f32: fp32 residual chain
// [B][T][C] rows are 64B-contiguous per 32-channel chunk => staging is pure
// 16B async copies; boundary rows redirect to a 64B zero block so every wave
// issues a uniform async count (deterministic s_wait_asynccnt bookkeeping).

typedef __attribute__((ext_vector_type(16))) _Float16 v16h;
typedef __attribute__((ext_vector_type(8)))  _Float16 v8h;
typedef __attribute__((ext_vector_type(8)))  float    v8f;

#define C_DIM 512
#define T_DIM 4096
#define B_DIM 8
#define KW    3
#define WELEM (C_DIM * C_DIM * KW) // 786432 elements per weight tensor

// Packed A-fragment weight layout: [k][chunk(16)][co16(32)][lane(32)][16 halfs]
#define PK_CO16_STRIDE  (32 * 16)              // 512
#define PK_CHUNK_STRIDE (32 * PK_CO16_STRIDE)  // 16384
#define PK_K_STRIDE     (16 * PK_CHUNK_STRIDE) // 262144 (== WELEM/3)

#define LSTR (138 * 32) // LDS halfs per buffer (max W = 128 + 2*5)

// ---------------------------------------------------------------------------
// s = mean(|w|): one block per tensor, deterministic fixed-order reduction.
__global__ __launch_bounds__(256) void absmean_kernel(const float* __restrict__ w1,
                                                      const float* __restrict__ w2,
                                                      float* __restrict__ scales) {
    __shared__ float red[256];
    const int t = blockIdx.x; // 0..5 : w1[0..2], w2[0..2]
    const float* src = (t < 3) ? (w1 + (size_t)t * WELEM)
                               : (w2 + (size_t)(t - 3) * WELEM);
    float s = 0.f;
    for (int i = threadIdx.x; i < WELEM; i += 256) s += fabsf(src[i]);
    red[threadIdx.x] = s;
    __syncthreads();
    for (int off = 128; off > 0; off >>= 1) {
        if (threadIdx.x < off) red[threadIdx.x] += red[threadIdx.x + off];
        __syncthreads();
    }
    if (threadIdx.x == 0) scales[t] = red[0] / (float)WELEM;
}

// ---------------------------------------------------------------------------
// Snake prep (exp() hoisted out of hot epilogues) + zero block for async
// boundary-row redirection.
__global__ __launch_bounds__(256) void prep_kernel(const float* __restrict__ alpha,
                                                   const float* __restrict__ beta,
                                                   float* __restrict__ aexp,
                                                   float* __restrict__ binv,
                                                   float* __restrict__ zerobuf) {
    const int i = blockIdx.x * 256 + threadIdx.x;
    if (i < 16) zerobuf[i] = 0.f;
    if (i < 3 * C_DIM) {
        aexp[i] = __expf(alpha[i]);
        binv[i] = 1.f / (__expf(beta[i]) + 1e-9f);
    }
}

// ---------------------------------------------------------------------------
// x f32 [B][C][T] -> xf16 [B][T][C] (one-time transpose; ~96MB traffic).
__global__ __launch_bounds__(256) void transpose_kernel(const float* __restrict__ x,
                                                        _Float16* __restrict__ xt) {
    __shared__ float tile[32][33];
    const int c0 = blockIdx.x * 32, t0 = blockIdx.y * 32, b = blockIdx.z;
    #pragma unroll
    for (int p = 0; p < 4; ++p) {
        const int idx = threadIdx.x + p * 256;
        const int c = idx >> 5, t = idx & 31; // consecutive tid -> consecutive t
        tile[c][t] = x[((size_t)(b * C_DIM + c0 + c)) * T_DIM + t0 + t];
    }
    __syncthreads();
    #pragma unroll
    for (int p = 0; p < 4; ++p) {
        const int idx = threadIdx.x + p * 256;
        const int t = idx >> 5, c = idx & 31; // consecutive tid -> consecutive c
        xt[((size_t)(b * T_DIM + t0 + t)) * C_DIM + c0 + c] = (_Float16)tile[c][t];
    }
}

// ---------------------------------------------------------------------------
// Ternarize + pack weights into the WMMA 16x32 f16 A-fragment layout
// (this code's contiguous k order, same as B's LDS reads): lane L covers M = L%16; its 16 halfs are K =
//   lanes 0-15 : [0..15],  lanes 16-31 : [16..31]  (per chunk)
__global__ __launch_bounds__(256) void pack_kernel(const float* __restrict__ w1,
                                                   const float* __restrict__ w2,
                                                   const float* __restrict__ scales,
                                                   _Float16* __restrict__ wp1,
                                                   _Float16* __restrict__ wp2) {
    long gid = (long)blockIdx.x * 256 + threadIdx.x;
    if (gid >= 6L * WELEM) return;
    const int t   = (int)(gid / WELEM);
    const int rem = (int)(gid % WELEM);
    const int k     = rem / PK_K_STRIDE;
    const int r2    = rem % PK_K_STRIDE;
    const int chunk = r2 / PK_CHUNK_STRIDE;
    const int r3    = r2 % PK_CHUNK_STRIDE;
    const int co16  = r3 / PK_CO16_STRIDE;
    const int r4    = r3 % PK_CO16_STRIDE;
    const int lane  = r4 / 16;
    const int h     = r4 % 16;

    const int m       = co16 * 16 + (lane & 15);
    const int halfsel = lane >> 4;
    const int g       = h >> 3;
    const int j       = h & 7;
    const int kk      = halfsel * 16 + g * 8 + j;
    const int ci      = chunk * 32 + kk;

    const float* src = (t < 3) ? (w1 + (size_t)t * WELEM)
                               : (w2 + (size_t)(t - 3) * WELEM);
    const float s = scales[t];
    const float v = src[((size_t)m * C_DIM + ci) * KW + k];
    float q = rintf(v / (s + 1e-5f)); // round-half-even like jnp.round
    q = fminf(1.f, fmaxf(-1.f, q));   // ternary; scale applied in epilogue
    _Float16* dst = (t < 3) ? (wp1 + (size_t)t * WELEM)
                            : (wp2 + (size_t)(t - 3) * WELEM);
    dst[rem] = (_Float16)q;
}

// ---------------------------------------------------------------------------
// Shared GEMM core: async-staged, double-buffered, 24 WMMA per cin-chunk.
// in16 is f16 [B][T][C]; block tile 128 Cout x 128 T; waves 4(M) x 2(N);
// wave tile 32 x 64 => acc[2][4]. Tap k shifts the LDS row by k*D.
template <int D>
__device__ __forceinline__ void conv_core(const _Float16* __restrict__ in16,
                                          const _Float16* __restrict__ wp,
                                          const float* __restrict__ zerobuf,
                                          _Float16* lds, int b, int cob, int t0,
                                          v8f (&acc)[2][4]) {
    constexpr int W    = 128 + 2 * D; // staged rows (with halo)
    constexpr int TOTB = W * 4;       // 16B copy tiles per chunk
    constexpr int NASY = (TOTB + 255) / 256;
    static_assert(NASY == 3, "s_wait_asynccnt literal assumes 3 issues/wave");

    const int tid    = threadIdx.x;
    const int lane   = tid & 31;
    const int wave   = tid >> 5;
    const int wm     = wave & 3;
    const int wn     = wave >> 2;
    const int nlane  = lane & 15;
    const int hsel16 = (lane >> 4) << 4;

    auto issue_async = [&](int chunk, int buf) {
        const int cib = chunk * 32;
        #pragma unroll
        for (int i = 0; i < NASY; ++i) {
            int ee = tid + i * 256;
            if (ee > TOTB - 1) ee = TOTB - 1; // clamp: duplicate copy, same data
            const int r = ee >> 2, q = ee & 3;
            const int tg = t0 - D + r;
            const char* gp;
            if ((unsigned)tg < (unsigned)T_DIM)
                gp = (const char*)(in16 + ((size_t)(b * T_DIM + tg)) * C_DIM + cib) + q * 16;
            else
                gp = (const char*)zerobuf + q * 16; // zero-pad halo rows
            unsigned ldsb = (unsigned)(size_t)(lds + buf * LSTR + r * 32) + q * 16;
            unsigned long long ga = (unsigned long long)(size_t)gp;
            asm volatile("global_load_async_to_lds_b128 %0, %1, off"
                         :: "v"(ldsb), "v"(ga) : "memory");
        }
    };

    issue_async(0, 0);
    for (int chunk = 0; chunk < 16; ++chunk) {
        if (chunk < 15) {
            issue_async(chunk + 1, (chunk + 1) & 1); // DMA overlaps this chunk's math
            asm volatile("s_wait_asynccnt 0x3" ::: "memory"); // chunk's copies landed
        } else {
            asm volatile("s_wait_asynccnt 0x0" ::: "memory");
        }
        __syncthreads(); // all waves' copies for buf[chunk&1] visible
        const int bufb = (chunk & 1) * LSTR;
        #pragma unroll
        for (int k = 0; k < 3; ++k) {
            v16h afrag[2];
            #pragma unroll
            for (int mf = 0; mf < 2; ++mf) {
                const int co16 = (cob + wm * 32 + mf * 16) >> 4;
                afrag[mf] = *(const v16h*)(wp + (size_t)k * PK_K_STRIDE +
                                           chunk * PK_CHUNK_STRIDE +
                                           co16 * PK_CO16_STRIDE + lane * 16);
            }
            #pragma unroll
            for (int nf = 0; nf < 4; ++nf) {
                const int row = wn * 64 + nf * 16 + nlane + k * D;
                const v16h bfrag = *(const v16h*)(lds + bufb + row * 32 + hsel16);
                #pragma unroll
                for (int mf = 0; mf < 2; ++mf) {
                    acc[mf][nf] = __builtin_amdgcn_wmma_f32_16x16x32_f16(
                        false, afrag[mf], false, bfrag, (short)0, acc[mf][nf],
                        false, false);
                }
            }
        }
        __syncthreads(); // reads of buf[chunk&1] done before it is re-filled
    }
}

// ---------------------------------------------------------------------------
// Conv1: h = snake(scale*conv(x16) + bias), h f16 [B][T][C].
template <int D>
__global__ __launch_bounds__(256) void bitconv1_kernel(
    const _Float16* __restrict__ in16, _Float16* __restrict__ hbuf,
    const _Float16* __restrict__ wp, const float* __restrict__ scales, int sidx,
    const float* __restrict__ bias, const float* __restrict__ aexp,
    const float* __restrict__ binv, const float* __restrict__ zerobuf) {
    __shared__ alignas(64) _Float16 lds[2 * LSTR];
    const int cob = blockIdx.x * 128, t0 = blockIdx.y * 128, b = blockIdx.z;
    v8f acc[2][4] = {};
    conv_core<D>(in16, wp, zerobuf, lds, b, cob, t0, acc);

    const int lane = threadIdx.x & 31, wave = threadIdx.x >> 5;
    const int wm = wave & 3, wn = wave >> 5 ? 0 : wave >> 2; // wn = wave>>2
    const int wn2 = wave >> 2;
    const float scale = scales[sidx];
    #pragma unroll
    for (int mf = 0; mf < 2; ++mf) {
        #pragma unroll
        for (int nf = 0; nf < 4; ++nf) {
            const int t   = t0 + wn2 * 64 + nf * 16 + (lane & 15);
            const int co0 = cob + wm * 32 + mf * 16 + ((lane >> 4) << 3);
            v8h hv;
            #pragma unroll
            for (int r = 0; r < 8; ++r) {
                const int co = co0 + r;
                float y = acc[mf][nf][r] * scale + bias[co];
                const float sn = __sinf(aexp[co] * y);
                y = y + sn * sn * binv[co];
                hv[r] = (_Float16)y;
            }
            *(v8h*)(hbuf + ((size_t)(b * T_DIM + t)) * C_DIM + co0) = hv;
        }
    }
    (void)wn;
}

// ---------------------------------------------------------------------------
// Conv2 (D=1): out = resid + scale*conv(h) + bias, f32; when OUT_CT==false it
// also emits the f16 [B][T][C] copy that feeds the next stage's conv1.
//   RES_CT : residual layout [B][C][T] (stage 0) vs [B][T][C]
//   OUT_CT : output layout   [B][C][T] (stage 2, d_out) vs [B][T][C]
template <bool RES_CT, bool OUT_CT>
__global__ __launch_bounds__(256) void bitconv2_kernel(
    const _Float16* __restrict__ hbuf, const float* __restrict__ resid,
    float* __restrict__ outp, _Float16* __restrict__ out16,
    const _Float16* __restrict__ wp, const float* __restrict__ scales, int sidx,
    const float* __restrict__ bias, const float* __restrict__ zerobuf) {
    __shared__ alignas(64) _Float16 lds[2 * LSTR];
    const int cob = blockIdx.x * 128, t0 = blockIdx.y * 128, b = blockIdx.z;
    v8f acc[2][4] = {};
    conv_core<1>(hbuf, wp, zerobuf, lds, b, cob, t0, acc);

    const int lane = threadIdx.x & 31, wave = threadIdx.x >> 5;
    const int wm = wave & 3, wn = wave >> 2;
    const float scale = scales[sidx];
    #pragma unroll
    for (int mf = 0; mf < 2; ++mf) {
        #pragma unroll
        for (int nf = 0; nf < 4; ++nf) {
            const int t   = t0 + wn * 64 + nf * 16 + (lane & 15);
            const int co0 = cob + wm * 32 + mf * 16 + ((lane >> 4) << 3);
            float o[8];
            #pragma unroll
            for (int r = 0; r < 8; ++r)
                o[r] = acc[mf][nf][r] * scale + bias[co0 + r];
            if (RES_CT) {
                #pragma unroll
                for (int r = 0; r < 8; ++r)
                    o[r] += resid[((size_t)(b * C_DIM + co0 + r)) * T_DIM + t];
            } else {
                const float4* rp =
                    (const float4*)(resid + ((size_t)(b * T_DIM + t)) * C_DIM + co0);
                const float4 ra = rp[0], rb = rp[1];
                o[0] += ra.x; o[1] += ra.y; o[2] += ra.z; o[3] += ra.w;
                o[4] += rb.x; o[5] += rb.y; o[6] += rb.z; o[7] += rb.w;
            }
            if (OUT_CT) {
                #pragma unroll
                for (int r = 0; r < 8; ++r)
                    outp[((size_t)(b * C_DIM + co0 + r)) * T_DIM + t] = o[r];
            } else {
                float4* op = (float4*)(outp + ((size_t)(b * T_DIM + t)) * C_DIM + co0);
                op[0] = make_float4(o[0], o[1], o[2], o[3]);
                op[1] = make_float4(o[4], o[5], o[6], o[7]);
                v8h hv;
                #pragma unroll
                for (int r = 0; r < 8; ++r) hv[r] = (_Float16)o[r];
                *(v8h*)(out16 + ((size_t)(b * T_DIM + t)) * C_DIM + co0) = hv;
            }
        }
    }
}

// ---------------------------------------------------------------------------
extern "C" void kernel_launch(void* const* d_in, const int* in_sizes, int n_in,
                              void* d_out, int out_size, void* d_ws, size_t ws_size,
                              hipStream_t stream) {
    const float* x  = (const float*)d_in[0];
    const float* w1 = (const float*)d_in[1];
    const float* b1 = (const float*)d_in[2];
    const float* al = (const float*)d_in[3];
    const float* be = (const float*)d_in[4];
    const float* w2 = (const float*)d_in[5];
    const float* b2 = (const float*)d_in[6];
    float* out = (float*)d_out;

    // Workspace (~138 MB):
    //   [0,64MB)    x2 fp32 [B][T][C]
    //   [64,96MB)   h f16 [B][T][C]
    //   [96,128MB)  xf16 [B][T][C] (x0T, then x1f16, then x2f16 -- reused)
    //   [128MB..)   packed ternary f16 weights (2 x 4.5MB), then
    //               zerobuf(64B) | aexp(6KB) | binv(6KB) | scales(24B)
    char* ws = (char*)d_ws;
    float*    xalt = (float*)ws;
    _Float16* hbuf = (_Float16*)(ws + ((size_t)64 << 20));
    _Float16* xf16 = (_Float16*)(ws + ((size_t)96 << 20));
    _Float16* wp1  = (_Float16*)(ws + ((size_t)128 << 20));
    _Float16* wp2  = wp1 + 3 * (size_t)WELEM;
    float* zerobuf = (float*)(wp2 + 3 * (size_t)WELEM);
    float* aexp    = zerobuf + 16;
    float* binv    = aexp + 3 * C_DIM;
    float* scales  = binv + 3 * C_DIM;

    absmean_kernel<<<6, 256, 0, stream>>>(w1, w2, scales);
    prep_kernel<<<6, 256, 0, stream>>>(al, be, aexp, binv, zerobuf);
    {
        long tot = 6L * WELEM;
        pack_kernel<<<(int)((tot + 255) / 256), 256, 0, stream>>>(w1, w2, scales, wp1, wp2);
    }
    transpose_kernel<<<dim3(C_DIM / 32, T_DIM / 32, B_DIM), 256, 0, stream>>>(x, xf16);

    dim3 grid(C_DIM / 128, T_DIM / 128, B_DIM), blk(256);

    // Stage 0 (d=1): xf16 -> h; x1 = x + conv2 -> d_out(TC) + xf16
    bitconv1_kernel<1><<<grid, blk, 0, stream>>>(xf16, hbuf,
        wp1 + 0 * (size_t)WELEM, scales, 0, b1 + 0, aexp + 0, binv + 0, zerobuf);
    bitconv2_kernel<true, false><<<grid, blk, 0, stream>>>(hbuf, x, out, xf16,
        wp2 + 0 * (size_t)WELEM, scales, 3, b2 + 0, zerobuf);

    // Stage 1 (d=3): xf16 -> h; x2 = x1 + conv2 -> xalt(TC) + xf16
    bitconv1_kernel<3><<<grid, blk, 0, stream>>>(xf16, hbuf,
        wp1 + 1 * (size_t)WELEM, scales, 1, b1 + C_DIM, aexp + C_DIM, binv + C_DIM, zerobuf);
    bitconv2_kernel<false, false><<<grid, blk, 0, stream>>>(hbuf, out, xalt, xf16,
        wp2 + 1 * (size_t)WELEM, scales, 4, b2 + C_DIM, zerobuf);

    // Stage 2 (d=5): xf16 -> h; final = x2 + conv2 -> d_out [B][C][T]
    bitconv1_kernel<5><<<grid, blk, 0, stream>>>(xf16, hbuf,
        wp1 + 2 * (size_t)WELEM, scales, 2, b1 + 2 * C_DIM, aexp + 2 * C_DIM, binv + 2 * C_DIM, zerobuf);
    bitconv2_kernel<false, true><<<grid, blk, 0, stream>>>(hbuf, xalt, out, nullptr,
        wp2 + 2 * (size_t)WELEM, scales, 5, b2 + 2 * C_DIM, zerobuf);
}
